// GNN_87471303950611
// MI455X (gfx1250) — matrix operand order repaired and checked by reading, not compiled, not run on hardware
//
#include <hip/hip_runtime.h>

// ---------------------------------------------------------------------------
// Problem constants (match reference)
// ---------------------------------------------------------------------------
#define N_NODES    50000
#define N_EDGES    800000
#define E_TOT      (N_EDGES + N_NODES)   // edges + self loops
#define IN_DIM     128
#define HID_DIM    96
#define OUT_DIM    96
#define NUM_GRAPHS 64
#define SLOPE_ATT  0.2f
#define SLOPE_ACT  0.01f

typedef __attribute__((ext_vector_type(16))) __bf16 v16bf;
typedef __attribute__((ext_vector_type(8)))  __bf16 v8bf;
typedef __attribute__((ext_vector_type(8)))  float  v8f;

// f32 -> bf16 round-to-nearest-even
__device__ __forceinline__ __bf16 f2bf(float f) {
    unsigned u = __float_as_uint(f);
    unsigned r = (u + 0x7FFFu + ((u >> 16) & 1u)) >> 16;
    unsigned short us = (unsigned short)r;
    return __builtin_bit_cast(__bf16, us);
}

// sign-aware float atomic max via integer atomics (L2 atomic units)
__device__ __forceinline__ void atomicMaxF32(float* addr, float val) {
    if (val >= 0.0f) {
        atomicMax((int*)addr, __float_as_int(val));
    } else {
        atomicMin((unsigned int*)addr, __float_as_uint(val));
    }
}

// ---------------------------------------------------------------------------
// f32 -> bf16 bulk convert (2 elements/thread, one dword store)
// ---------------------------------------------------------------------------
__global__ void cvt_f32_bf16(const float* __restrict__ src,
                             __bf16* __restrict__ dst, int n2 /* = n/2 */) {
    int t = blockIdx.x * blockDim.x + threadIdx.x;
    if (t >= n2) return;
    __bf16 a = f2bf(src[2 * t]);
    __bf16 b = f2bf(src[2 * t + 1]);
    unsigned short ua = __builtin_bit_cast(unsigned short, a);
    unsigned short ub = __builtin_bit_cast(unsigned short, b);
    ((unsigned*)dst)[t] = (unsigned)ua | ((unsigned)ub << 16);
}

// ---------------------------------------------------------------------------
// Pre-swizzle W[K,D] (f32, row-major) into per-lane WMMA B fragments (bf16).
//   B 32x16 layout: lane = (half<<4)|l : col n = n0+l, elem q: K = (q>>3)*16+half*8+(q&7)
//   Wpk[((tileN*KB + kb)*32 + lane)*16 + q] = bf16(W[(kb*32+(q>>3)*16+half*8+(q&7))*D + n0+l])
// Tiny: (D/16)*(K/32)*32 fragments of 16 bf16.
// ---------------------------------------------------------------------------
__global__ void pack_W_frags(const float* __restrict__ W,
                             __bf16* __restrict__ Wpk, int K, int D) {
    const int KB = K / 32;
    const int nFrag = (D / 16) * KB * 32;
    int t = blockIdx.x * blockDim.x + threadIdx.x;
    if (t >= nFrag) return;
    int lane  = t & 31;
    int kb    = (t >> 5) % KB;
    int tileN = (t >> 5) / KB;
    int half  = lane >> 4;
    int l     = lane & 15;
    __bf16* out = Wpk + (size_t)t * 16;
#pragma unroll
    for (int q = 0; q < 16; ++q) {            // q -> K offset (q>>3)*16+half*8+(q&7) within 32
        int k = kb * 32 + (q >> 3) * 16 + half * 8 + (q & 7);
        out[q] = f2bf(W[(size_t)k * D + tileN * 16 + l]);
    }
}

// ---------------------------------------------------------------------------
// H[N,D] = Xbf[N,K] @ W (pre-packed fragments), K fully unrolled.
// One 16x16 tile per wave, 8 waves per block along M.
// Per K-step: 2x b128 A loads (contiguous per lane) + 2x b128 B loads + WMMA.
// ---------------------------------------------------------------------------
template <int K>
__global__ void gat_gemm_wmma(const __bf16* __restrict__ Xbf,
                              const __bf16* __restrict__ Wpk,
                              float* __restrict__ H, int N, int D) {
    constexpr int KB = K / 32;
    const int lane  = threadIdx.x & 31;
    const int wave  = threadIdx.x >> 5;
    const int tileM = blockIdx.x * 8 + wave;
    if (tileM >= (N >> 4)) return;
    const int tileN = blockIdx.y;
    const int m0 = tileM << 4;
    const int n0 = tileN << 4;
    const int half = lane >> 4;
    const int l    = lane & 15;

    const __bf16* xr = Xbf + (size_t)(m0 + l) * K;
    const __bf16* wf = Wpk + ((size_t)tileN * KB * 32 + lane) * 16;

    v8f acc = {};
#pragma unroll
    for (int kb = 0; kb < KB; ++kb) {
        const int kk = kb * 32;
        // A 16x32 fragment: two contiguous 8-element (16B) runs per lane
        v8bf a0 = *(const v8bf*)(xr + kk + half * 8);
        v8bf a1 = *(const v8bf*)(xr + kk + 16 + half * 8);
        v16bf a = __builtin_shufflevector(a0, a1, 0, 1, 2, 3, 4, 5, 6, 7,
                                                  8, 9, 10, 11, 12, 13, 14, 15);
        // B fragment: 32B contiguous per lane
        v16bf b = *(const v16bf*)(wf + (size_t)kb * 32 * 16);
        acc = __builtin_amdgcn_wmma_f32_16x16x32_bf16(
            false, a, false, b, (short)0, acc, false, false);
    }

#pragma unroll
    for (int i = 0; i < 8; ++i) {
        H[(size_t)(m0 + i + 8 * half) * D + (n0 + l)] = acc[i];
    }
}

// ---------------------------------------------------------------------------
// Per-node attention scalars: as[n]=h[n].a_src, ad[n]=h[n].a_dst (wave/node)
// ---------------------------------------------------------------------------
__global__ void gat_alphas(const float* __restrict__ H,
                           const float* __restrict__ a_src,
                           const float* __restrict__ a_dst,
                           float* __restrict__ as, float* __restrict__ ad,
                           int N, int D) {
    int node = blockIdx.x * (blockDim.x >> 5) + (threadIdx.x >> 5);
    int lane = threadIdx.x & 31;
    if (node >= N) return;
    const float* h = H + (size_t)node * D;
    float s = 0.f, d = 0.f;
#pragma unroll
    for (int j = 0; j < 3; ++j) {            // D = 96 = 3*32
        float hv = h[lane + 32 * j];
        s += hv * a_src[lane + 32 * j];
        d += hv * a_dst[lane + 32 * j];
    }
#pragma unroll
    for (int off = 16; off > 0; off >>= 1) {
        s += __shfl_down(s, off, 32);
        d += __shfl_down(d, off, 32);
    }
    if (lane == 0) { as[node] = s; ad[node] = d; }
}

__global__ void fill_f32(float* __restrict__ p, float v, int n) {
    int t = blockIdx.x * blockDim.x + threadIdx.x;
    if (t < n) p[t] = v;
}

// e = leaky_relu(as[src]+ad[dst], 0.2); m[dst] = max(m[dst], e)
__global__ void gat_edge_max(const int* __restrict__ ei,
                             const float* __restrict__ as,
                             const float* __restrict__ ad,
                             float* __restrict__ e, float* __restrict__ m) {
    int t = blockIdx.x * blockDim.x + threadIdx.x;
    if (t >= E_TOT) return;
    int s, d;
    if (t < N_EDGES) { s = ei[t]; d = ei[N_EDGES + t]; }
    else             { s = d = t - N_EDGES; }
    float v = as[s] + ad[d];
    v = (v > 0.f) ? v : SLOPE_ATT * v;
    e[t] = v;
    atomicMaxF32(&m[d], v);
}

// e <- exp(e - m[dst]); denom[dst] += e
__global__ void gat_edge_expsum(const int* __restrict__ ei,
                                float* __restrict__ e,
                                const float* __restrict__ m,
                                float* __restrict__ denom) {
    int t = blockIdx.x * blockDim.x + threadIdx.x;
    if (t >= E_TOT) return;
    int d = (t < N_EDGES) ? ei[N_EDGES + t] : (t - N_EDGES);
    float ex = __expf(e[t] - m[d]);
    e[t] = ex;
    atomicAdd(&denom[d], ex);
}

// out[dst] += (e/denom[dst]) * h[src] ; one wave per edge, coalesced dims
__global__ void gat_edge_agg(const int* __restrict__ ei,
                             const float* __restrict__ e,
                             const float* __restrict__ denom,
                             const float* __restrict__ H,
                             float* __restrict__ out, int D) {
    int w    = blockIdx.x * (blockDim.x >> 5) + (threadIdx.x >> 5);
    int lane = threadIdx.x & 31;
    if (w >= E_TOT) return;
    int s, d;
    if (w < N_EDGES) { s = ei[w]; d = ei[N_EDGES + w]; }
    else             { s = d = w - N_EDGES; }
    const float* hs = H + (size_t)s * D;
    // hide the random-gather latency: each lane prefetches its element's line
    __builtin_prefetch(hs + lane, 0, 0);
    float alpha = e[w] / (denom[d] + 1e-16f);
    float* od = out + (size_t)d * D;
#pragma unroll
    for (int j = 0; j < 3; ++j) {            // D = 96 = 3*32 -> coalesced
        atomicAdd(&od[lane + 32 * j], alpha * hs[lane + 32 * j]);
    }
}

// out += bias ; optional leaky_relu(0.01)
__global__ void gat_bias_act(float* __restrict__ out, const float* __restrict__ b,
                             int N, int D, int do_act) {
    int t = blockIdx.x * blockDim.x + threadIdx.x;
    if (t >= N * D) return;
    float v = out[t] + b[t % D];
    if (do_act) v = (v > 0.f) ? v : SLOPE_ACT * v;
    out[t] = v;
}

// graph pooling: sums[g,:] += h[n,:], counts[g] += 1  (wave per node)
__global__ void gat_pool(const float* __restrict__ h, const int* __restrict__ batch,
                         float* __restrict__ sums, float* __restrict__ counts, int D) {
    int node = blockIdx.x * (blockDim.x >> 5) + (threadIdx.x >> 5);
    int lane = threadIdx.x & 31;
    if (node >= N_NODES) return;
    int g = batch[node];
    const float* hn = h + (size_t)node * D;
    float* sg = sums + (size_t)g * D;
#pragma unroll
    for (int j = 0; j < 3; ++j) {
        atomicAdd(&sg[lane + 32 * j], hn[lane + 32 * j]);
    }
    if (lane == 0) atomicAdd(&counts[g], 1.0f);
}

__global__ void gat_pool_div(float* __restrict__ out, const float* __restrict__ sums,
                             const float* __restrict__ counts, int D) {
    int t = blockIdx.x * blockDim.x + threadIdx.x;
    if (t >= NUM_GRAPHS * D) return;
    float c = counts[t / D];
    out[t] = sums[t] / fmaxf(c, 1.0f);
}

// ---------------------------------------------------------------------------
// Host-side orchestration
// ---------------------------------------------------------------------------
static inline int ceil_div(int a, int b) { return (a + b - 1) / b; }

template <int K>
static void run_gat_layer(const float* X,
                          const float* W, const float* a_src, const float* a_dst,
                          const float* bias,
                          __bf16* Xbf, __bf16* Wpk,
                          float* H, float* AGG, float* AS, float* AD,
                          float* M, float* DEN, float* E,
                          const int* ei, int do_act, hipStream_t stream) {
    const int D = HID_DIM;  // all layer outputs are 96
    // 0) precision staging: X -> bf16, W -> packed bf16 fragments
    cvt_f32_bf16<<<ceil_div(N_NODES * K / 2, 256), 256, 0, stream>>>(X, Xbf, N_NODES * K / 2);
    pack_W_frags<<<ceil_div((D / 16) * (K / 32) * 32, 256), 256, 0, stream>>>(W, Wpk, K, D);
    // 1) H = X @ W  (WMMA, K fully unrolled)
    {
        dim3 grid(ceil_div(N_NODES / 16, 8), D / 16);
        gat_gemm_wmma<K><<<grid, 256, 0, stream>>>(Xbf, Wpk, H, N_NODES, D);
    }
    // 2) attention scalars
    gat_alphas<<<ceil_div(N_NODES, 8), 256, 0, stream>>>(H, a_src, a_dst, AS, AD, N_NODES, D);
    // 3) init segment-softmax state
    fill_f32<<<ceil_div(N_NODES, 256), 256, 0, stream>>>(M, -__builtin_inff(), N_NODES);
    fill_f32<<<ceil_div(N_NODES, 256), 256, 0, stream>>>(DEN, 0.0f, N_NODES);
    // 4) per-edge logit + segment max
    gat_edge_max<<<ceil_div(E_TOT, 256), 256, 0, stream>>>(ei, AS, AD, E, M);
    // 5) exp + segment sum
    gat_edge_expsum<<<ceil_div(E_TOT, 256), 256, 0, stream>>>(ei, E, M, DEN);
    // 6) zero aggregation buffer
    fill_f32<<<ceil_div(N_NODES * D, 256), 256, 0, stream>>>(AGG, 0.0f, N_NODES * D);
    // 7) weighted message aggregation (wave per edge)
    gat_edge_agg<<<ceil_div(E_TOT, 8), 256, 0, stream>>>(ei, E, DEN, H, AGG, D);
    // 8) bias (+ leaky relu between layers)
    gat_bias_act<<<ceil_div(N_NODES * D, 256), 256, 0, stream>>>(AGG, bias, N_NODES, D, do_act);
}

extern "C" void kernel_launch(void* const* d_in, const int* in_sizes, int n_in,
                              void* d_out, int out_size, void* d_ws, size_t ws_size,
                              hipStream_t stream) {
    (void)in_sizes; (void)n_in; (void)out_size; (void)ws_size;

    const float* x      = (const float*)d_in[0];
    const int*   ei     = (const int*)  d_in[1];   // [2, N_EDGES] flat
    const int*   batch  = (const int*)  d_in[2];
    const float* W1     = (const float*)d_in[3];
    const float* asrc1  = (const float*)d_in[4];
    const float* adst1  = (const float*)d_in[5];
    const float* b1     = (const float*)d_in[6];
    const float* W2     = (const float*)d_in[7];
    const float* asrc2  = (const float*)d_in[8];
    const float* adst2  = (const float*)d_in[9];
    const float* b2     = (const float*)d_in[10];
    const float* W3     = (const float*)d_in[11];
    const float* asrc3  = (const float*)d_in[12];
    const float* adst3  = (const float*)d_in[13];
    const float* b3     = (const float*)d_in[14];
    float* out = (float*)d_out;

    // -------- workspace carve --------
    float* ws   = (float*)d_ws;
    float* H    = ws;                                // N*96
    float* AGG  = H    + (size_t)N_NODES * HID_DIM;  // N*96
    float* AS   = AGG  + (size_t)N_NODES * HID_DIM;  // N
    float* AD   = AS   + N_NODES;                    // N
    float* M    = AD   + N_NODES;                    // N
    float* DEN  = M    + N_NODES;                    // N
    float* E    = DEN  + N_NODES;                    // E_TOT
    float* SUMS = E    + E_TOT;                      // G*96
    float* CNT  = SUMS + NUM_GRAPHS * OUT_DIM;       // G
    // bf16 region (32B aligned)
    uintptr_t p = (uintptr_t)(CNT + NUM_GRAPHS);
    p = (p + 31u) & ~(uintptr_t)31u;
    __bf16* Xbf = (__bf16*)p;                        // N * max(K) = N*128
    p += (size_t)N_NODES * IN_DIM * sizeof(__bf16);
    p = (p + 31u) & ~(uintptr_t)31u;
    __bf16* Wpk = (__bf16*)p;                        // <= 6*4*32*16 bf16 (24 KB)

    // layer 1: X = x (K=128)
    run_gat_layer<IN_DIM>(x, W1, asrc1, adst1, b1, Xbf, Wpk,
                          H, AGG, AS, AD, M, DEN, E, ei, 1, stream);
    // layer 2: X = AGG (K=96)  (cvt reads AGG fully before it is re-zeroed)
    run_gat_layer<HID_DIM>(AGG, W2, asrc2, adst2, b2, Xbf, Wpk,
                           H, AGG, AS, AD, M, DEN, E, ei, 1, stream);
    // layer 3: no activation
    run_gat_layer<HID_DIM>(AGG, W3, asrc3, adst3, b3, Xbf, Wpk,
                           H, AGG, AS, AD, M, DEN, E, ei, 0, stream);

    // global mean pool
    fill_f32<<<ceil_div(NUM_GRAPHS * OUT_DIM, 256), 256, 0, stream>>>(SUMS, 0.0f, NUM_GRAPHS * OUT_DIM);
    fill_f32<<<1, 256, 0, stream>>>(CNT, 0.0f, NUM_GRAPHS);
    gat_pool<<<ceil_div(N_NODES, 8), 256, 0, stream>>>(AGG, batch, SUMS, CNT, OUT_DIM);
    gat_pool_div<<<ceil_div(NUM_GRAPHS * OUT_DIM, 256), 256, 0, stream>>>(out, SUMS, CNT, OUT_DIM);
}
